// MaskedSelfAttention_48533130445269
// MI455X (gfx1250) — matrix operand order repaired and checked by reading, not compiled, not run on hardware
//
#include <hip/hip_runtime.h>
#include <hip/hip_bf16.h>
#include <math.h>

typedef __bf16 bf16_t;
typedef __attribute__((ext_vector_type(16))) __bf16 v16bf;
typedef __attribute__((ext_vector_type(8)))  __bf16 v8bf;
typedef __attribute__((ext_vector_type(4)))  __bf16 v4bf;
typedef __attribute__((ext_vector_type(8)))  float  v8f;
typedef __attribute__((ext_vector_type(4)))  float  v4f;

#define SEQ_T   2048
#define CDIM    1024
#define NHEADS  16
#define DHEAD   64
#define NTOK    8192          // B*T

// ---------------------------------------------------------------------------
// WMMA helper (CDNA5 bf16 16x16x32, fp32 accumulate)
// ---------------------------------------------------------------------------
__device__ __forceinline__ v8f wmma_bf16(v16bf a, v16bf b, v8f c) {
    return __builtin_amdgcn_wmma_f32_16x16x32_bf16(
        /*neg_a=*/false, a, /*neg_b=*/false, b,
        /*c_mod=*/(short)0, c, /*reuse_a=*/false, /*reuse_b=*/false);
}

// A-matrix 16x32 bf16 fragment from row-major [M][K] (stride ld elems).
// ISA layout: lane m=lane&15; lanes<16 hold K {0..7,16..23}; lanes>=16 hold
// K {8..15,24..31}  -> two contiguous 16B chunks at cols c and c+16.
__device__ __forceinline__ v16bf load_frag_a(const bf16_t* p, int ld, int lane) {
    int m = lane & 15;
    int c = (lane < 16) ? 0 : 8;
    v8bf lo = *(const v8bf*)(p + (size_t)m * ld + c);
    v8bf hi = *(const v8bf*)(p + (size_t)m * ld + c + 16);
    return __builtin_shufflevector(lo, hi, 0,1,2,3,4,5,6,7,8,9,10,11,12,13,14,15);
}

// B-matrix 32x16 bf16 fragment from K-major Bt[N][K] (stride ld elems).
// ISA layout: lane n=lane&15; lanes<16 hold K {0..7,16..23}; lanes>=16 hold
// K {8..15,24..31}  -> two contiguous 16B chunks at cols base and base+16.
__device__ __forceinline__ v16bf load_frag_b(const bf16_t* p, int ld, int lane) {
    int n    = lane & 15;
    int base = (lane < 16) ? 0 : 8;
    v8bf lo = *(const v8bf*)(p + (size_t)n * ld + base);
    v8bf hi = *(const v8bf*)(p + (size_t)n * ld + base + 16);
    return __builtin_shufflevector(lo, hi, 0,1,2,3,4,5,6,7,8,9,10,11,12,13,14,15);
}

// ---------------------------------------------------------------------------
// Weight transpose + fp32->bf16 convert:  out[Cc][R] = bf16(in[R][Cc])
// ---------------------------------------------------------------------------
__global__ __launch_bounds__(256)
void k_transpose_cvt(const float* __restrict__ in, bf16_t* __restrict__ out,
                     int R, int Cc) {
    __shared__ float tile[32][33];
    int bx = blockIdx.x * 32, by = blockIdx.y * 32;
    int tx = threadIdx.x, ty = threadIdx.y;   // (32, 8)
#pragma unroll
    for (int k = 0; k < 4; ++k)
        tile[ty + k * 8][tx] = in[(size_t)(by + ty + k * 8) * Cc + bx + tx];
    __syncthreads();
#pragma unroll
    for (int k = 0; k < 4; ++k)
        out[(size_t)(bx + ty + k * 8) * R + by + tx] = (bf16_t)tile[tx][ty + k * 8];
}

// ---------------------------------------------------------------------------
// Bulk fp32 -> bf16
// ---------------------------------------------------------------------------
__global__ __launch_bounds__(256)
void k_cvt_bf16(const float* __restrict__ in, bf16_t* __restrict__ out, size_t n) {
    size_t i = ((size_t)blockIdx.x * blockDim.x + threadIdx.x) * 4;
    if (i + 3 < n) {
        v4f d = *(const v4f*)(in + i);
        v4bf o;
#pragma unroll
        for (int j = 0; j < 4; ++j) o[j] = (bf16_t)d[j];
        *(v4bf*)(out + i) = o;
    }
}

// ---------------------------------------------------------------------------
// GEMM: Y[row][o] = sum_i A[row][i] * Wt[o][i] + bias[o]
// A: bf16 [NTOK][CDIM] row-major.  Wt: bf16 [CDIM][CDIM] (W transposed).
// mode 0: Q  -> bf16 [b][h][t][d], scaled by 1/sqrt(D)
// mode 1: K  -> bf16 [b][h][t][d]
// mode 2: V  -> bf16 [b][h][d][t]   (transposed for PV B-operand)
// mode 3: out-> f32  [row][o]       (final projection)
// Workgroup: 256 thr (8 waves), block tile BM=128 x BN=128, BK=32 via LDS.
// Wave tile: 32x64 -> 2 A-frags x 4 B-frags = 8 WMMA per K-step.
// Register-staged double buffering: global loads for tile k+1 issue before
// the WMMA chain of tile k, hiding L2/HBM latency under matrix ops.
// ---------------------------------------------------------------------------
#define BM 128
#define BN 128
#define BK 32
#define LDA 40
#define LDB 40

__global__ __launch_bounds__(256)
void k_gemm(const bf16_t* __restrict__ A, const bf16_t* __restrict__ Wt,
            const float* __restrict__ bias, void* __restrict__ Out, int mode) {
    __shared__ alignas(16) bf16_t As[BM * LDA];   // 10 KiB
    __shared__ alignas(16) bf16_t Bs[BN * LDB];   // 10 KiB

    int tid  = threadIdx.x;
    int wave = tid >> 5;
    int lane = tid & 31;
    int wm = (wave & 3) * 32;     // wave M offset within block tile (0..96)
    int wn = (wave >> 2) * 64;    // wave N offset within block tile (0/64)
    int m0 = blockIdx.y * BM;
    int n0 = blockIdx.x * BN;

    int ldr = tid >> 2;           // 0..63 staging row base
    int ldc = (tid & 3) * 8;      // 0/8/16/24 staging col

    const bf16_t* gA0 = A  + (size_t)(m0 + ldr)      * CDIM + ldc;
    const bf16_t* gA1 = A  + (size_t)(m0 + ldr + 64) * CDIM + ldc;
    const bf16_t* gB0 = Wt + (size_t)(n0 + ldr)      * CDIM + ldc;
    const bf16_t* gB1 = Wt + (size_t)(n0 + ldr + 64) * CDIM + ldc;

    v8f acc[2][4] = {};

    // prologue: stage tile k0=0
    v8bf ra0 = *(const v8bf*)(gA0);
    v8bf ra1 = *(const v8bf*)(gA1);
    v8bf rb0 = *(const v8bf*)(gB0);
    v8bf rb1 = *(const v8bf*)(gB1);
    *(v8bf*)(As + ldr * LDA + ldc)        = ra0;
    *(v8bf*)(As + (ldr + 64) * LDA + ldc) = ra1;
    *(v8bf*)(Bs + ldr * LDB + ldc)        = rb0;
    *(v8bf*)(Bs + (ldr + 64) * LDB + ldc) = rb1;
    __syncthreads();

    for (int k0 = 0; k0 < CDIM; k0 += BK) {
        int kn = k0 + BK;
        if (kn < CDIM) {              // issue next tile's loads early
            ra0 = *(const v8bf*)(gA0 + kn);
            ra1 = *(const v8bf*)(gA1 + kn);
            rb0 = *(const v8bf*)(gB0 + kn);
            rb1 = *(const v8bf*)(gB1 + kn);
            if (kn + BK < CDIM) {     // warm caches two steps ahead
                __builtin_prefetch(gA0 + kn + BK, 0, 3);
                __builtin_prefetch(gB0 + kn + BK, 0, 3);
            }
        }

        v16bf a0 = load_frag_a(As + (wm +  0) * LDA, LDA, lane);
        v16bf a1 = load_frag_a(As + (wm + 16) * LDA, LDA, lane);
#pragma unroll
        for (int j = 0; j < 4; ++j) {
            v16bf b = load_frag_b(Bs + (wn + j * 16) * LDB, LDB, lane);
            acc[0][j] = wmma_bf16(a0, b, acc[0][j]);
            acc[1][j] = wmma_bf16(a1, b, acc[1][j]);
        }

        __syncthreads();              // all waves done reading this tile
        if (kn < CDIM) {
            *(v8bf*)(As + ldr * LDA + ldc)        = ra0;
            *(v8bf*)(As + (ldr + 64) * LDA + ldc) = ra1;
            *(v8bf*)(Bs + ldr * LDB + ldc)        = rb0;
            *(v8bf*)(Bs + (ldr + 64) * LDB + ldc) = rb1;
        }
        __syncthreads();
    }

    // epilogue: D-fragment layout -> element (r,lane): M=r+(lane<16?0:8), N=lane&15
    int half = lane >> 4;
    int ln   = lane & 15;
#pragma unroll
    for (int i = 0; i < 2; ++i) {
#pragma unroll
        for (int j = 0; j < 4; ++j) {
            int col  = n0 + wn + j * 16 + ln;
            float bv = bias[col];
#pragma unroll
            for (int r = 0; r < 8; ++r) {
                int row = m0 + wm + i * 16 + r + half * 8;
                float v = acc[i][j][r] + bv;
                if (mode == 3) {
                    ((float*)Out)[(size_t)row * CDIM + col] = v;
                } else {
                    int bb = row >> 11, t = row & (SEQ_T - 1);
                    int hh = col >> 6,  d = col & (DHEAD - 1);
                    if (mode == 0) {
                        v *= 0.125f;  // 1/sqrt(64)
                        ((bf16_t*)Out)[(((size_t)(bb * NHEADS + hh)) * SEQ_T + t) * DHEAD + d] = (bf16_t)v;
                    } else if (mode == 1) {
                        ((bf16_t*)Out)[(((size_t)(bb * NHEADS + hh)) * SEQ_T + t) * DHEAD + d] = (bf16_t)v;
                    } else {
                        ((bf16_t*)Out)[(((size_t)(bb * NHEADS + hh)) * DHEAD + d) * SEQ_T + t] = (bf16_t)v;
                    }
                }
            }
        }
    }
}

// ---------------------------------------------------------------------------
// Causal flash attention: one wave per (b, h, 32-query tile) -> two 16-row
// M-tiles sharing every K/V fragment load.
// Q,K: bf16 [b][h][t][d]; Vt: bf16 [b][h][d][t]; ctx out: bf16 [b*T][C].
// ---------------------------------------------------------------------------
__global__ __launch_bounds__(32)
void k_attn(const bf16_t* __restrict__ Qb, const bf16_t* __restrict__ Kb,
            const bf16_t* __restrict__ Vt, bf16_t* __restrict__ ctx) {
    __shared__ alignas(16) bf16_t Pt[2][16 * 40];   // P tiles 16x32, padded

    int lane = threadIdx.x;
    int bid  = blockIdx.x;
    int qt   = bid & 63;           // T/32 = 64 tiles
    int bh   = bid >> 6;           // b*NHEADS + h
    int q0   = qt * 32;

    const bf16_t* Qh = Qb + (size_t)bh * SEQ_T * DHEAD;
    const bf16_t* Kh = Kb + (size_t)bh * SEQ_T * DHEAD;
    const bf16_t* Vh = Vt + (size_t)bh * DHEAD * SEQ_T;

    v16bf qa[2][2];
#pragma unroll
    for (int i = 0; i < 2; ++i) {
        qa[i][0] = load_frag_a(Qh + (size_t)(q0 + i * 16) * DHEAD +  0, DHEAD, lane);
        qa[i][1] = load_frag_a(Qh + (size_t)(q0 + i * 16) * DHEAD + 32, DHEAD, lane);
    }

    v8f acc[2][4] = {};
    float m_r[2][8], l_r[2][8];
#pragma unroll
    for (int i = 0; i < 2; ++i)
#pragma unroll
        for (int r = 0; r < 8; ++r) { m_r[i][r] = -INFINITY; l_r[i][r] = 0.f; }

    int half = lane >> 4;
    int ln   = lane & 15;
    int nblk = q0 / 32 + 1;        // keys needed: 0 .. q0+31

    for (int blk = 0; blk < nblk; ++blk) {
        int s0 = blk * 32;
#pragma unroll
        for (int sub = 0; sub < 2; ++sub) {
            int sb = s0 + sub * 16;
            // K fragments shared by both query tiles
            v16bf kb0 = load_frag_b(Kh + (size_t)sb * DHEAD +  0, DHEAD, lane);
            v16bf kb1 = load_frag_b(Kh + (size_t)sb * DHEAD + 32, DHEAD, lane);
#pragma unroll
            for (int i = 0; i < 2; ++i) {
                int qbase = q0 + i * 16;
                if (sb > qbase + 15) {      // subtile fully masked for this tile
#pragma unroll
                    for (int r = 0; r < 8; ++r)
                        Pt[i][(r + half * 8) * 40 + sub * 16 + ln] = (bf16_t)0.f;
                    continue;
                }
                v8f s = {};
                s = wmma_bf16(qa[i][0], kb0, s);
                s = wmma_bf16(qa[i][1], kb1, s);

                int scol = sb + ln;
                float pv[8], mt[8];
#pragma unroll
                for (int r = 0; r < 8; ++r) {
                    int qrow = qbase + r + half * 8;
                    float x = s[r];
                    if (scol > qrow) x = -INFINITY;   // causal mask
                    pv[r] = x;
                }
#pragma unroll
                for (int r = 0; r < 8; ++r) {         // row max over 16 lanes
                    float v = pv[r];
                    v = fmaxf(v, __shfl_xor(v, 1));
                    v = fmaxf(v, __shfl_xor(v, 2));
                    v = fmaxf(v, __shfl_xor(v, 4));
                    v = fmaxf(v, __shfl_xor(v, 8));
                    mt[r] = v;
                }
#pragma unroll
                for (int r = 0; r < 8; ++r) {
                    float mn    = fmaxf(m_r[i][r], mt[r]);
                    float alpha = __expf(m_r[i][r] - mn);
                    float p     = __expf(pv[r] - mn);
                    float rs = p;
                    rs += __shfl_xor(rs, 1);
                    rs += __shfl_xor(rs, 2);
                    rs += __shfl_xor(rs, 4);
                    rs += __shfl_xor(rs, 8);
                    l_r[i][r] = l_r[i][r] * alpha + rs;
                    m_r[i][r] = mn;
                    acc[i][0][r] *= alpha; acc[i][1][r] *= alpha;
                    acc[i][2][r] *= alpha; acc[i][3][r] *= alpha;
                    Pt[i][(r + half * 8) * 40 + sub * 16 + ln] = (bf16_t)p;
                }
            }
        }
        __syncthreads();   // single wave: orders LDS P stores before frag loads
        v16bf pa0 = load_frag_a(Pt[0], 40, lane);
        v16bf pa1 = load_frag_a(Pt[1], 40, lane);
#pragma unroll
        for (int j = 0; j < 4; ++j) {   // V fragment shared by both tiles
            v16bf vb = load_frag_b(Vh + (size_t)(j * 16) * SEQ_T + s0, SEQ_T, lane);
            acc[0][j] = wmma_bf16(pa0, vb, acc[0][j]);
            acc[1][j] = wmma_bf16(pa1, vb, acc[1][j]);
        }
        __syncthreads();
    }

    int b = bh >> 4, h = bh & 15;
#pragma unroll
    for (int i = 0; i < 2; ++i) {
#pragma unroll
        for (int r = 0; r < 8; ++r) {
            float inv = 1.0f / l_r[i][r];
            int t = q0 + i * 16 + r + half * 8;
            size_t rowbase = ((size_t)b * SEQ_T + t) * CDIM + h * DHEAD;
            ctx[rowbase +  0 + ln] = (bf16_t)(acc[i][0][r] * inv);
            ctx[rowbase + 16 + ln] = (bf16_t)(acc[i][1][r] * inv);
            ctx[rowbase + 32 + ln] = (bf16_t)(acc[i][2][r] * inv);
            ctx[rowbase + 48 + ln] = (bf16_t)(acc[i][3][r] * inv);
        }
    }
}

// ---------------------------------------------------------------------------
// Host-side orchestration
// ---------------------------------------------------------------------------
extern "C" void kernel_launch(void* const* d_in, const int* in_sizes, int n_in,
                              void* d_out, int out_size, void* d_ws, size_t ws_size,
                              hipStream_t stream) {
    (void)in_sizes; (void)n_in; (void)out_size; (void)ws_size;

    const float* x  = (const float*)d_in[0];
    const float* Wq = (const float*)d_in[1];
    const float* bq = (const float*)d_in[2];
    const float* Wk = (const float*)d_in[3];
    const float* bk = (const float*)d_in[4];
    const float* Wv = (const float*)d_in[5];
    const float* bv = (const float*)d_in[6];
    const float* Wo = (const float*)d_in[7];
    const float* bo = (const float*)d_in[8];

    char* ws = (char*)d_ws;
    const size_t MiB = 1024 * 1024;
    bf16_t* Wq_t = (bf16_t*)(ws + 0 * MiB);
    bf16_t* Wk_t = (bf16_t*)(ws + 2 * MiB);
    bf16_t* Wv_t = (bf16_t*)(ws + 4 * MiB);
    bf16_t* Wo_t = (bf16_t*)(ws + 6 * MiB);
    bf16_t* xb   = (bf16_t*)(ws + 8 * MiB);    // [NTOK][CDIM]
    bf16_t* Qb   = (bf16_t*)(ws + 24 * MiB);   // [b][h][t][d]
    bf16_t* Kb   = (bf16_t*)(ws + 40 * MiB);   // [b][h][t][d]
    bf16_t* Vtb  = (bf16_t*)(ws + 56 * MiB);   // [b][h][d][t]
    bf16_t* ctxb = (bf16_t*)(ws + 72 * MiB);   // [NTOK][CDIM]

    dim3 tb(32, 8);
    dim3 tg(CDIM / 32, CDIM / 32);
    k_transpose_cvt<<<tg, tb, 0, stream>>>(Wq, Wq_t, CDIM, CDIM);
    k_transpose_cvt<<<tg, tb, 0, stream>>>(Wk, Wk_t, CDIM, CDIM);
    k_transpose_cvt<<<tg, tb, 0, stream>>>(Wv, Wv_t, CDIM, CDIM);
    k_transpose_cvt<<<tg, tb, 0, stream>>>(Wo, Wo_t, CDIM, CDIM);

    size_t nx = (size_t)NTOK * CDIM;
    k_cvt_bf16<<<(unsigned)(nx / (4 * 256)), 256, 0, stream>>>(x, xb, nx);

    dim3 gg(CDIM / BN, NTOK / BM);   // (8, 64)
    k_gemm<<<gg, 256, 0, stream>>>(xb, Wq_t, bq, (void*)Qb, 0);
    k_gemm<<<gg, 256, 0, stream>>>(xb, Wk_t, bk, (void*)Kb, 1);
    k_gemm<<<gg, 256, 0, stream>>>(xb, Wv_t, bv, (void*)Vtb, 2);

    k_attn<<<4 * NHEADS * (SEQ_T / 32), 32, 0, stream>>>(Qb, Kb, Vtb, ctxb);

    k_gemm<<<gg, 256, 0, stream>>>(ctxb, Wo_t, bo, d_out, 3);
}
